// SelfAttention_171798691943
// MI455X (gfx1250) — matrix operand order repaired and checked by reading, not compiled, not run on hardware
//
#include <hip/hip_runtime.h>
#include <cstdint>

// ---------------- CDNA5 wave32 WMMA types ----------------
typedef __attribute__((ext_vector_type(16))) __bf16 v16bf;
typedef __attribute__((ext_vector_type(4)))  __bf16 v4bf;
typedef __attribute__((ext_vector_type(8)))  float  v8f;

static __device__ __forceinline__ v8f wmma_bf16(v16bf a, v16bf b, v8f c) {
  return __builtin_amdgcn_wmma_f32_16x16x32_bf16(false, a, false, b, (short)0, c,
                                                 false, false);
}

// Async global->LDS copy, 16B per lane (512B per wave-instruction). ASYNCcnt.
static __device__ __forceinline__ void async_b128(const void* lds,
                                                  const void* gaddr) {
  const uint32_t l = (uint32_t)(uintptr_t)lds;  // LDS byte offset (low 32 bits)
  asm volatile("global_load_async_to_lds_b128 %0, %1, off" ::"v"(l), "v"(gaddr)
               : "memory");
}
#define WAIT_ASYNC(n) asm volatile("s_wait_asynccnt " #n ::: "memory")

// A-matrix 16x32 bf16 fragment (ISA 7.12.2): lane l holds row M=l&15;
// lanes 0-15: elems 0..7 -> K=0..7, 8..15 -> K=16..23
// lanes 16-31: elems 0..7 -> K=8..15, 8..15 -> K=24..31
static __device__ __forceinline__ v16bf ld_frag_A(const __bf16* base, int ld) {
  const int lane = threadIdx.x & 31;
  const int row  = lane & 15;
  const int kb   = (lane >> 4) << 3;  // 0 or 8 elems
  const __bf16* p = base + (size_t)row * ld;
  union { v16bf v; uint4 q[2]; } u;
  u.q[0] = *(const uint4*)(p + kb);
  u.q[1] = *(const uint4*)(p + 16 + kb);
  return u.v;
}

// B-matrix 32x16 bf16 fragment: lane l holds column N=l&15;
// lanes 0-15: elems 0..7 -> K=0..7, 8..15 -> K=16..23; lanes 16-31: K=8..15, K=24..31.
// LDS layout assumed [n][k] with k contiguous (ld elems per n-row).
static __device__ __forceinline__ v16bf ld_frag_B(const __bf16* base, int ld) {
  const int lane = threadIdx.x & 31;
  const int row  = lane & 15;         // N index
  const int kb   = (lane >> 4) << 3;  // 0 or 8 elems
  const __bf16* p = base + (size_t)row * ld + kb;
  union { v16bf v; uint4 q[2]; } u;
  u.q[0] = *(const uint4*)(p);
  u.q[1] = *(const uint4*)(p + 16);
  return u.v;
}

// ---------------- problem constants ----------------
#define BB   4
#define SS   2048
#define EE   2048
#define HH   16
#define DH   128
#define SKIPN 16
// 1/sqrt(128)
#define SCALE 0.08838834764831845f
// -ln(10000)/PAIRS
#define NEG_LOG1E4_OVER_16 (-0.5756462732485115f)

// ---------------- f32 -> bf16 convert ----------------
__global__ __launch_bounds__(256) void cvt_f32_bf16(const float* __restrict__ s,
                                                    __bf16* __restrict__ d, int n4) {
  int i = blockIdx.x * 256 + threadIdx.x;
  if (i < n4) {
    const float4 v = ((const float4*)s)[i];
    v4bf o;
    o[0] = (__bf16)v.x; o[1] = (__bf16)v.y; o[2] = (__bf16)v.z; o[3] = (__bf16)v.w;
    ((v4bf*)d)[i] = o;
  }
}

// ---------------- bf16 GEMM: C[m][n] = sum_k A[m][k]*W[n][k] ----------------
// block: 256 thr (8 waves). block tile 128(M) x 64(N), K-slice 64 (2 WMMA
// k-steps), double-buffered LDS fed by global_load_async_to_lds_b128.
// wave w: rows mBase+w*16..+15, all 64 cols (4 C tiles) -> 8 WMMA / stage.
// EPI 0: store fp32 C.  EPI 1: fused RoPE + split into q/k/v^T bf16.
template <int EPI>
__global__ __launch_bounds__(256) void gemm_bf16_k(
    const __bf16* __restrict__ A, const __bf16* __restrict__ W,
    int M, int N, int K,
    float* __restrict__ Cout, const int* __restrict__ pos,
    __bf16* __restrict__ qO, __bf16* __restrict__ kO, __bf16* __restrict__ vtO) {
  __shared__ __bf16 As[2][128][72];  // 128 x 64 (+8 pad) x2 buffers
  __shared__ __bf16 Bs[2][64][72];   // 64  x 64 (+8 pad) x2 buffers
  const int tid  = threadIdx.x;
  const int wave = tid >> 5, lane = tid & 31, hi = lane >> 4, ln = lane & 15;
  const int mBase = blockIdx.y * 128;
  const int nBase = blockIdx.x * 64;

  // 6 async b128 per thread per stage (4 for A, 2 for B)
  auto issue = [&](int kb, int buf) {
#pragma unroll
    for (int i = 0; i < 4; ++i) {
      int c = i * 256 + tid;           // A: 1024 chunks of 8 elems (8 per row)
      int row = c >> 3, col = (c & 7) * 8;
      async_b128(&As[buf][row][col],
                 A + (size_t)(mBase + row) * K + kb + col);
    }
#pragma unroll
    for (int i = 0; i < 2; ++i) {
      int c = i * 256 + tid;           // B: 512 chunks
      int row = c >> 3, col = (c & 7) * 8;
      async_b128(&Bs[buf][row][col],
                 W + (size_t)(nBase + row) * K + kb + col);
    }
  };

  v8f acc[4] = {};
  const int nSl = K / 64;
  issue(0, 0);
  for (int s = 0; s < nSl; ++s) {
    const int buf = s & 1;
    __syncthreads();  // buffer buf^1 free (stage s-1 consumed by all waves)
    if (s + 1 < nSl) {
      issue((s + 1) * 64, buf ^ 1);
      WAIT_ASYNC(0x6);  // keep only the 6 newest in flight -> stage s landed
    } else {
      WAIT_ASYNC(0x0);
    }
    __syncthreads();  // stage s visible to every wave
#pragma unroll
    for (int ks = 0; ks < 2; ++ks) {
      v16bf af = ld_frag_A(&As[buf][wave * 16][ks * 32], 72);
#pragma unroll
      for (int c = 0; c < 4; ++c) {
        v16bf bf_ = ld_frag_B(&Bs[buf][c * 16][ks * 32], 72);
        acc[c] = wmma_bf16(af, bf_, acc[c]);
      }
    }
  }
  const int mt = mBase + wave * 16;
  if constexpr (EPI == 0) {
    (void)pos; (void)qO; (void)kO; (void)vtO;
#pragma unroll
    for (int c = 0; c < 4; ++c) {
      int n = nBase + c * 16 + ln;
#pragma unroll
      for (int r = 0; r < 8; ++r) {
        int m = mt + r + 8 * hi;
        Cout[(size_t)m * N + n] = acc[c][r];
      }
    }
  } else {
    (void)Cout;
    // acc pairs (0,1) and (2,3) are the (half0,half1) RoPE partners:
    // d = axis*32 + half*16 + p, tiles are 16 cols so each tile = one half.
#pragma unroll
    for (int pr = 0; pr < 2; ++pr) {
      int n0    = nBase + pr * 32 + ln;   // col of half0 element
      int which = n0 >> 11;               // 0=q 1=k 2=v
      int rem   = n0 & (EE - 1);
      int hh    = rem >> 7;               // head
      int d0    = rem & (DH - 1);
      int pp    = d0 & 15;                // pair index
      int axis  = d0 >> 5;
      float invf = __expf(NEG_LOG1E4_OVER_16 * (float)pp);
#pragma unroll
      for (int r = 0; r < 8; ++r) {
        int m  = mt + r + 8 * hi;
        int bI = m >> 11;            // /S
        int sI = m & (SS - 1);
        float v0 = acc[pr * 2][r];
        float v1 = acc[pr * 2 + 1][r];
        if (which < 2 && sI >= SKIPN) {
          float ang = (float)pos[((size_t)bI * SS + sI) * 4 + axis] * invf;
          float sn, cs;
          __sincosf(ang, &sn, &cs);
          float o0 = v0 * cs - v1 * sn;
          float o1 = v0 * sn + v1 * cs;
          v0 = o0; v1 = o1;
        }
        size_t hb = (size_t)(bI * HH + hh);
        if (which == 0) {
          size_t idx = (hb * SS + sI) * DH + d0;
          qO[idx] = (__bf16)v0; qO[idx + 16] = (__bf16)v1;
        } else if (which == 1) {
          size_t idx = (hb * SS + sI) * DH + d0;
          kO[idx] = (__bf16)v0; kO[idx + 16] = (__bf16)v1;
        } else {  // V stored transposed: vt[b][h][d][s]
          size_t idx = (hb * DH + d0) * SS + sI;
          vtO[idx] = (__bf16)v0; vtO[idx + (size_t)16 * SS] = (__bf16)v1;
        }
      }
    }
  }
}

// ---------------- flash attention ----------------
// grid: (S/128, B*H), block 256. wave w owns 16 q rows; K/V^T double-buffered
// in LDS via async copies, 32-key chunks; online softmax; P transposed
// through per-wave LDS scratch (same-wave LDS ops are in-order, no barrier).
__global__ __launch_bounds__(256) void attn_k(const __bf16* __restrict__ Q,
                                              const __bf16* __restrict__ Kt,
                                              const __bf16* __restrict__ Vt,
                                              __bf16* __restrict__ Oout) {
  __shared__ __bf16 Ks[2][32][136];   // 32 keys x 128 d (+pad) x2
  __shared__ __bf16 Vs[2][128][40];   // 128 d x 32 keys (+pad) x2
  __shared__ __bf16 Ps[8][16][40];    // per-wave P scratch 16q x 32key (+pad)
  const int tid  = threadIdx.x;
  const int wave = tid >> 5, lane = tid & 31, hi = lane >> 4, ln = lane & 15;
  const int bh    = blockIdx.y;  // b*H + h
  const int qRow0 = blockIdx.x * 128 + wave * 16;
  const size_t bhS = (size_t)bh * SS;
  const float NEG_INF = -__builtin_inff();

  // 4 async b128 per thread per chunk (2 for K, 2 for V^T)
  auto issue = [&](int kt, int buf) {
#pragma unroll
    for (int i = 0; i < 2; ++i) {
      int c = i * 256 + tid;
      {
        int key = c >> 4, dc = (c & 15) * 8;  // K: 32 rows x 256B
        async_b128(&Ks[buf][key][dc],
                   Kt + (bhS + (size_t)kt * 32 + key) * DH + dc);
      }
      {
        int dd = c >> 2, ko = (c & 3) * 8;    // V^T: 128 rows x 64B
        async_b128(&Vs[buf][dd][ko],
                   Vt + ((size_t)bh * DH + dd) * SS + kt * 32 + ko);
      }
    }
  };

  const __bf16* qbase = Q + (bhS + qRow0) * DH;
  v16bf qf[4];
#pragma unroll
  for (int j = 0; j < 4; ++j) qf[j] = ld_frag_A(qbase + j * 32, DH);

  v8f O[8] = {};
  float m_i[8], l_i[8];
#pragma unroll
  for (int r = 0; r < 8; ++r) { m_i[r] = NEG_INF; l_i[r] = 0.f; }

  const int nCh = blockIdx.x * 4 + 4;  // causal: keys < qBlockEnd
  issue(0, 0);
  for (int kt = 0; kt < nCh; ++kt) {
    const int buf = kt & 1;
    __syncthreads();  // buffer buf^1 free (chunk kt-1 consumed by all waves)
    if (kt + 1 < nCh) {
      issue(kt + 1, buf ^ 1);
      WAIT_ASYNC(0x4);  // only newest 4 in flight -> chunk kt landed
    } else {
      WAIT_ASYNC(0x0);
    }
    __syncthreads();  // chunk kt visible to every wave

    // scores: two 16x16 key tiles, accumulate over Dh in 4 WMMA k-steps
    v8f s0{}, s1{};
#pragma unroll
    for (int j = 0; j < 4; ++j) {
      v16bf b0 = ld_frag_B(&Ks[buf][0][j * 32], 136);
      v16bf b1 = ld_frag_B(&Ks[buf][16][j * 32], 136);
      s0 = wmma_bf16(qf[j], b0, s0);
      s1 = wmma_bf16(qf[j], b1, s1);
    }

    // online softmax (rows striped: row = r + 8*hi, col = ln in 16-lane group)
    float a_r[8];
#pragma unroll
    for (int r = 0; r < 8; ++r) {
      int qrow = qRow0 + r + 8 * hi;
      int key0 = kt * 32 + ln;
      float v0 = s0[r] * SCALE;
      float v1 = s1[r] * SCALE;
      if (key0 > qrow) v0 = NEG_INF;
      if (key0 + 16 > qrow) v1 = NEG_INF;
      float mx = fmaxf(v0, v1);
      mx = fmaxf(mx, __shfl_xor(mx, 1, 32));
      mx = fmaxf(mx, __shfl_xor(mx, 2, 32));
      mx = fmaxf(mx, __shfl_xor(mx, 4, 32));
      mx = fmaxf(mx, __shfl_xor(mx, 8, 32));
      float mnew = fmaxf(m_i[r], mx);
      float alpha, p0, p1;
      if (mnew == NEG_INF) {  // fully masked so far
        alpha = 1.f; p0 = 0.f; p1 = 0.f;
      } else {
        alpha = __expf(m_i[r] - mnew);
        p0 = (v0 == NEG_INF) ? 0.f : __expf(v0 - mnew);
        p1 = (v1 == NEG_INF) ? 0.f : __expf(v1 - mnew);
      }
      float rs = p0 + p1;
      rs += __shfl_xor(rs, 1, 32);
      rs += __shfl_xor(rs, 2, 32);
      rs += __shfl_xor(rs, 4, 32);
      rs += __shfl_xor(rs, 8, 32);
      l_i[r] = l_i[r] * alpha + rs;
      m_i[r] = mnew;
      a_r[r] = alpha;
      Ps[wave][r + 8 * hi][ln]      = (__bf16)p0;
      Ps[wave][r + 8 * hi][16 + ln] = (__bf16)p1;
    }
#pragma unroll
    for (int dt = 0; dt < 8; ++dt)
#pragma unroll
      for (int r = 0; r < 8; ++r) O[dt][r] *= a_r[r];

    // Same-wave LDS store->load is in order (DScnt, ISA 7.3): no barrier.
    v16bf pa = ld_frag_A(&Ps[wave][0][0], 40);
#pragma unroll
    for (int dt = 0; dt < 8; ++dt) {
      v16bf vb = ld_frag_B(&Vs[buf][dt * 16][0], 40);
      O[dt] = wmma_bf16(pa, vb, O[dt]);
    }
  }

  // finalize: O /= l, store bf16 to attn[b][s][h*Dh+d]
  const int bI = bh / HH, hI = bh % HH;
#pragma unroll
  for (int r = 0; r < 8; ++r) {
    float inv = 1.f / l_i[r];
    size_t rowb =
        ((size_t)bI * SS + qRow0 + r + 8 * hi) * EE + (size_t)hI * DH;
#pragma unroll
    for (int dt = 0; dt < 8; ++dt)
      Oout[rowb + dt * 16 + ln] = (__bf16)(O[dt][r] * inv);
  }
}

// ---------------- launcher ----------------
extern "C" void kernel_launch(void* const* d_in, const int* in_sizes, int n_in,
                              void* d_out, int out_size, void* d_ws, size_t ws_size,
                              hipStream_t stream) {
  (void)in_sizes; (void)n_in; (void)out_size; (void)ws_size;
  const float* x    = (const float*)d_in[0];
  // d_in[1] attention_mask: all-true, causal mask dominates -> unused
  const int*   pos  = (const int*)d_in[2];
  const float* Wqkv = (const float*)d_in[3];
  const float* Wout = (const float*)d_in[4];
  float* out = (float*)d_out;

  const size_t nX    = (size_t)BB * SS * EE;      // 16,777,216
  const size_t nWqkv = (size_t)3 * EE * EE;       // 12,582,912
  const size_t nWout = (size_t)EE * EE;           //  4,194,304

  char* w = (char*)d_ws;
  __bf16* x_bf    = (__bf16*)w; w += nX * 2;
  __bf16* wqkv_bf = (__bf16*)w; w += nWqkv * 2;
  __bf16* wout_bf = (__bf16*)w; w += nWout * 2;
  __bf16* q_bf    = (__bf16*)w; w += nX * 2;      // [B,H,S,Dh]
  __bf16* k_bf    = (__bf16*)w; w += nX * 2;      // [B,H,S,Dh]
  __bf16* vt_bf   = (__bf16*)w; w += nX * 2;      // [B,H,Dh,S]
  __bf16* attn_bf = (__bf16*)w; w += nX * 2;      // [B,S,E]

  cvt_f32_bf16<<<(int)(nX / 4 / 256), 256, 0, stream>>>(x, x_bf, (int)(nX / 4));
  cvt_f32_bf16<<<(int)(nWqkv / 4 / 256), 256, 0, stream>>>(Wqkv, wqkv_bf,
                                                           (int)(nWqkv / 4));
  cvt_f32_bf16<<<(int)(nWout / 4 / 256), 256, 0, stream>>>(Wout, wout_bf,
                                                           (int)(nWout / 4));

  // QKV projection + fused RoPE -> q, k, v^T (bf16)
  {
    dim3 grid((3 * EE) / 64, (BB * SS) / 128);
    gemm_bf16_k<1><<<grid, 256, 0, stream>>>(x_bf, wqkv_bf, BB * SS, 3 * EE, EE,
                                             nullptr, pos, q_bf, k_bf, vt_bf);
  }

  // flash attention
  {
    dim3 grid(SS / 128, BB * HH);
    attn_k<<<grid, 256, 0, stream>>>(q_bf, k_bf, vt_bf, attn_bf);
  }

  // output projection -> fp32
  {
    dim3 grid(EE / 64, (BB * SS) / 128);
    gemm_bf16_k<0><<<grid, 256, 0, stream>>>(attn_bf, wout_bf, BB * SS, EE, EE,
                                             out, nullptr, nullptr, nullptr,
                                             nullptr);
  }
}
